// Attention_43963285242463
// MI455X (gfx1250) — matrix operand order repaired and checked by reading, not compiled, not run on hardware
//
#include <hip/hip_runtime.h>
#include <hip/hip_bf16.h>
#include <math.h>
#include <stddef.h>

// ---------------- problem constants ----------------
#define HIDDEN  1280
#define NHEADS  10
#define HDIM    128
#define BATCH   4
#define SEQ     2048
#define NROWS   (BATCH * SEQ)           // 8192 rows for the projections
// softmax scale (1/sqrt(128)) folded with log2(e) so we can use exp2f
#define SM_C    0.12751744f

#if defined(__has_builtin) && __has_builtin(__builtin_amdgcn_sched_barrier)
#define SCHED_FENCE() __builtin_amdgcn_sched_barrier(0)
#else
#define SCHED_FENCE() ((void)0)
#endif

typedef __bf16 v16bf __attribute__((ext_vector_type(16)));
typedef float  v8f   __attribute__((ext_vector_type(8)));
typedef unsigned int u32x4 __attribute__((ext_vector_type(4)));
typedef int          i32x4 __attribute__((ext_vector_type(4)));
typedef int          i32x8 __attribute__((ext_vector_type(8)));

// ---------------- small helpers ----------------
static __device__ __forceinline__ unsigned short f2bf(float f) {
  unsigned int u = __float_as_uint(f);
  u += 0x7fffu + ((u >> 16) & 1u);          // round-to-nearest-even
  return (unsigned short)(u >> 16);
}
static __device__ __forceinline__ float bf2f(unsigned short s) {
  return __uint_as_float(((unsigned int)s) << 16);
}

union FragU { uint4 u[2]; v16bf v; };

// Load 16 bf16 as two 16-byte chunks (chunk addresses must be 16B aligned).
static __device__ __forceinline__ v16bf load_frag_pair(const unsigned short* p0,
                                                       const unsigned short* p1) {
  FragU r;
  r.u[0] = *(const uint4*)p0;
  r.u[1] = *(const uint4*)p1;
  return r.v;
}

static __device__ __forceinline__ v8f wmma_bf16(v16bf a, v16bf b, v8f c) {
  // (neg_a, A, neg_b, B, c_mod, C, reuse_a, reuse_b)
  return __builtin_amdgcn_wmma_f32_16x16x32_bf16(false, a, false, b, (short)0, c,
                                                 false, false);
}

// ---------------- TDM: 2-D bf16 tile -> LDS (ISA ch.8 D#, 6-arg builtin) ----
// Tensor: rows of `tdim0` elements, row stride `stride0` elements (2B each).
// Tile: tile1 rows x tile0 elements, landing row-major at LDS byte `lds_off`.
static __device__ __forceinline__ void tdm_load_2d(const void* gptr,
                                                   unsigned lds_off,
                                                   unsigned tdim0, unsigned tdim1,
                                                   unsigned stride0,
                                                   unsigned tile0, unsigned tile1) {
  unsigned long long ga = (unsigned long long)(size_t)gptr;
  u32x4 g0;
  g0[0] = 1u;                                              // count=1 (valid D#)
  g0[1] = lds_off;                                         // lds_addr
  g0[2] = (unsigned)ga;                                    // global_addr[31:0]
  g0[3] = (unsigned)((ga >> 32) & 0x01FFFFFFu) | (2u << 30); // addr[56:32]|type=2
  i32x8 g1;
  g1[0] = (int)(1u << 16);                                 // data_size = 2 bytes
  g1[1] = (int)((tdim0 & 0xFFFFu) << 16);                  // tensor_dim0[15:0]
  g1[2] = (int)(((tdim0 >> 16) & 0xFFFFu) | ((tdim1 & 0xFFFFu) << 16));
  g1[3] = (int)(((tdim1 >> 16) & 0xFFFFu) | ((tile0 & 0xFFFFu) << 16));
  g1[4] = (int)(tile1 & 0xFFFFu);                          // tile_dim1 (tile_dim2=0)
  g1[5] = (int)stride0;                                    // tensor_dim0_stride[31:0]
  g1[6] = 0;
  g1[7] = 0;
  i32x4 z4 = {0, 0, 0, 0};                                 // groups 2/3 unused (2-D)
  i32x8 z8 = {0, 0, 0, 0, 0, 0, 0, 0};
  __builtin_amdgcn_tensor_load_to_lds(g0, g1, z4, z4, z8, 0);
}

// ---------------- fp32 -> bf16 convert ----------------
__global__ void k_f32_to_bf16(const float* __restrict__ in,
                              unsigned short* __restrict__ out, int n) {
  int i = blockIdx.x * blockDim.x + threadIdx.x;
  if (i < n) out[i] = f2bf(in[i]);
}

// ---------------- WMMA GEMM: C(MxN) = A(MxK) @ W(NxK)^T ----------------
// Block = 256 threads (8 waves). Block tile = 64 rows x 128 cols.
// Wave (w) tile: rows 16*(w>>1), cols 64*(w&1): four 16x16 accumulators.
// A-fragment interleave per ISA 7.12.2 (16-bit A 16x32):
//   lane<16 : row=lane,   K = {k0..k0+7, k0+16..k0+23}
//   lane>=16: row=lane-16,K = {k0+8..k0+15, k0+24..k0+31}
// B-fragment (32x16, column n = W row n, same K interleave as A):
//   lane<16 : col=lane, K = {k0..k0+7, k0+16..k0+23} ; lane>=16: col=lane-16, K = {k0+8..k0+15, k0+24..k0+31}
template <typename OutT>
__global__ void __launch_bounds__(256)
k_gemm_bf16(const unsigned short* __restrict__ A,
            const unsigned short* __restrict__ W,
            OutT* __restrict__ C, int M, int N, int K) {
  const int lane = threadIdx.x & 31;
  const int wave = threadIdx.x >> 5;
  const int m0 = blockIdx.x * 64  + (wave >> 1) * 16;
  const int n0 = blockIdx.y * 128 + (wave & 1) * 64;

  v8f acc[4] = {};
  const unsigned short* Arow =
      A + (size_t)(m0 + (lane & 15)) * K + ((lane < 16) ? 0 : 8);
  const int colK = (lane & 15);
  const int bko  = (lane < 16) ? 0 : 8;

  for (int k0 = 0; k0 < K; k0 += 32) {
    if (k0 + 32 < K) {
      __builtin_prefetch(Arow + k0 + 32, 0, 0);
      __builtin_prefetch(W + (size_t)(n0 + colK) * K + k0 + 32, 0, 0);
    }
    // batch all loads first so they clause and overlap the previous WMMAs
    v16bf a = load_frag_pair(Arow + k0, Arow + k0 + 16);
    v16bf b[4];
#pragma unroll
    for (int j = 0; j < 4; ++j) {
      const unsigned short* Bp = W + (size_t)(n0 + j * 16 + colK) * K + k0 + bko;
      b[j] = load_frag_pair(Bp, Bp + 16);
    }
    SCHED_FENCE();   // keep the load clause ahead of the WMMA burst
#pragma unroll
    for (int j = 0; j < 4; ++j) acc[j] = wmma_bf16(a, b[j], acc[j]);
  }
  // C/D layout: VGPR i -> row i (lanes 0-15) / row i+8 (lanes 16-31); N = lane&15
#pragma unroll
  for (int j = 0; j < 4; ++j) {
    int col = n0 + j * 16 + (lane & 15);
#pragma unroll
    for (int i = 0; i < 8; ++i) {
      int row = m0 + i + ((lane < 16) ? 0 : 8);
      float v = acc[j][i];
      if constexpr (sizeof(OutT) == 2)
        ((unsigned short*)C)[(size_t)row * N + col] = f2bf(v);
      else
        ((float*)C)[(size_t)row * N + col] = v;
    }
  }
}

// ---------------- RoPE + RMS-norm + V transpose ----------------
__global__ void __launch_bounds__(128)
k_rope_rms(const unsigned short* __restrict__ Qr,
           const unsigned short* __restrict__ Kr,
           const unsigned short* __restrict__ Vr,
           unsigned short* __restrict__ Qb,
           unsigned short* __restrict__ Kb,
           unsigned short* __restrict__ Vt) {
  const int rid = blockIdx.x;                 // b*SEQ*NHEADS + t*NHEADS + h
  const int h = rid % NHEADS;
  const int t = (rid / NHEADS) % SEQ;
  const int b = rid / (NHEADS * SEQ);
  const int d = threadIdx.x;                  // 0..127
  const size_t src = ((size_t)(b * SEQ + t)) * HIDDEN + h * HDIM;

  const int p = d & 63;
  float fr = -__expf((float)p * 0.14391435f); // ln(10000)/64
  float ang = (float)t * fr;
  float cs = __cosf(ang), sn = __sinf(ang);

  float q_self = bf2f(Qr[src + d]);
  float q_pair = bf2f(Qr[src + (d ^ 64)]);
  float k_self = bf2f(Kr[src + d]);
  float k_pair = bf2f(Kr[src + (d ^ 64)]);
  float rq = (d < 64) ? (q_self * cs - q_pair * sn) : (q_self * cs + q_pair * sn);
  float rk = (d < 64) ? (k_self * cs - k_pair * sn) : (k_self * cs + k_pair * sn);

  float sq = rq * rq, sk = rk * rk;
#pragma unroll
  for (int m = 16; m; m >>= 1) {
    sq += __shfl_xor(sq, m, 32);
    sk += __shfl_xor(sk, m, 32);
  }
  __shared__ float redq[4], redk[4];
  int w = threadIdx.x >> 5;
  if ((threadIdx.x & 31) == 0) { redq[w] = sq; redk[w] = sk; }
  __syncthreads();
  float tq = redq[0] + redq[1] + redq[2] + redq[3];
  float tk = redk[0] + redk[1] + redk[2] + redk[3];
  float rms_q = rsqrtf(tq * (1.0f / HDIM) + 1e-5f);
  float rms_k = rsqrtf(tk * (1.0f / HDIM) + 1e-5f);

  const size_t bh = (size_t)(b * NHEADS + h);
  Qb[(bh * SEQ + t) * HDIM + d] = f2bf(rq * rms_q);
  Kb[(bh * SEQ + t) * HDIM + d] = f2bf(rk * rms_k);
  Vt[bh * (size_t)HDIM * SEQ + (size_t)d * SEQ + t] = Vr[src + d];
}

// ---------------- flash attention with TDM-staged K/V tiles ----------------
// Grid: (SEQ/128, B*NHEADS). Block = 256 threads (8 waves); each wave owns a
// 16-query tile and a 16x128 f32 accumulator. Per key-tile of 32:
//   - wave0 DMAs next K-tile (32x128) and V-tile (128x32, from Vt) into the
//     spare LDS buffer via TENSOR_LOAD_TO_LDS, waits tensorcnt<=2 so the
//     current tile is resident, then the block barriers and computes.
struct SmemAttn {
  unsigned short kbuf[2][32 * HDIM];   // 2 x 8 KB  (row = key, col = d)
  unsigned short vbuf[2][HDIM * 32];   // 2 x 8 KB  (row = d,   col = key)
  unsigned short p[8][16 * 32];        // 8 KB per-wave P staging
};

__global__ void __launch_bounds__(256)
k_attn(const unsigned short* __restrict__ Qb,
       const unsigned short* __restrict__ Kb,
       const unsigned short* __restrict__ Vt,
       unsigned short* __restrict__ Out) {
  __shared__ __align__(16) SmemAttn smem;

  const int lane = threadIdx.x & 31;
  const int wave = threadIdx.x >> 5;
  const int bh = blockIdx.y;                       // 0..B*NHEADS-1
  const int q0 = blockIdx.x * 128 + wave * 16;

  const unsigned short* Q  = Qb + ((size_t)bh * SEQ + q0) * HDIM;
  const unsigned short* Kp = Kb + (size_t)bh * SEQ * HDIM;
  const unsigned short* Vp = Vt + (size_t)bh * HDIM * SEQ;

  // Q A-fragments for each 32-wide slice of d
  v16bf qf[4];
  {
    const unsigned short* Ar =
        Q + (size_t)(lane & 15) * HDIM + ((lane < 16) ? 0 : 8);
#pragma unroll
    for (int dt = 0; dt < 4; ++dt)
      qf[dt] = load_frag_pair(Ar + dt * 32, Ar + dt * 32 + 16);
  }

  v8f acc[8] = {};
  float mrow[8], lrow[8];
#pragma unroll
  for (int i = 0; i < 8; ++i) { mrow[i] = -INFINITY; lrow[i] = 0.0f; }

  const int colK = lane & 15;
  const int bko  = (lane < 16) ? 0 : 8;

  const unsigned offK0 = (unsigned)offsetof(SmemAttn, kbuf);
  const unsigned offV0 = (unsigned)offsetof(SmemAttn, vbuf);
  const unsigned kbytes = 32 * HDIM * 2;   // 8192

  if (threadIdx.x == 0) {                  // prologue: DMA tile 0 into buffer 0
    tdm_load_2d(Kp, offK0, HDIM, 32, HDIM, HDIM, 32);
    tdm_load_2d(Vp, offV0, 32, HDIM, SEQ, 32, HDIM);
  }

  const int NT = SEQ / 32;
  for (int kt = 0; kt < NT; ++kt) {
    const int key0 = kt * 32;
    const int buf  = kt & 1;
    if (threadIdx.x == 0) {
      if (kt + 1 < NT) {   // DMA next tile into the spare buffer, then wait for
        tdm_load_2d(Kp + (size_t)(key0 + 32) * HDIM, offK0 + (buf ^ 1) * kbytes,
                    HDIM, 32, HDIM, HDIM, 32);
        tdm_load_2d(Vp + (key0 + 32), offV0 + (buf ^ 1) * kbytes,
                    32, HDIM, SEQ, 32, HDIM);
        __builtin_amdgcn_s_wait_tensorcnt(2);   // ... the current tile only
      } else {
        __builtin_amdgcn_s_wait_tensorcnt(0);
      }
    }
    __syncthreads();                      // current tile resident for all waves
    const unsigned short* kb = smem.kbuf[buf];
    const unsigned short* vb = smem.vbuf[buf];

    // scores from LDS: batch all 8 K-fragments first so the ds_loads clause
    // and overlap, then burst the 8 WMMAs.
    v8f s0 = {}, s1 = {};
    {
      const unsigned short* B0 = kb + (size_t)colK * HDIM + bko;
      const unsigned short* B1 = B0 + 16 * HDIM;
      v16bf bk[8];
#pragma unroll
      for (int dt = 0; dt < 4; ++dt) {
        bk[dt]     = load_frag_pair(B0 + dt * 32, B0 + dt * 32 + 16);
        bk[4 + dt] = load_frag_pair(B1 + dt * 32, B1 + dt * 32 + 16);
      }
      SCHED_FENCE();   // keep the 16 ds_loads ahead of the WMMA burst
#pragma unroll
      for (int dt = 0; dt < 4; ++dt) {
        s0 = wmma_bf16(qf[dt], bk[dt], s0);
        s1 = wmma_bf16(qf[dt], bk[4 + dt], s1);
      }
    }
    // online softmax. C-layout: elem i -> row i (lanes<16) / row i+8 (lanes>=16)
    float pr0[8], pr1[8], fsc[8];
#pragma unroll
    for (int i = 0; i < 8; ++i) {
      float smax = fmaxf(s0[i], s1[i]);
#pragma unroll
      for (int m = 8; m; m >>= 1) smax = fmaxf(smax, __shfl_xor(smax, m, 16));
      float mnew = fmaxf(mrow[i], smax);
      float f  = exp2f((mrow[i] - mnew) * SM_C);
      float p0 = exp2f((s0[i] - mnew) * SM_C);
      float p1 = exp2f((s1[i] - mnew) * SM_C);
      float rs = p0 + p1;
#pragma unroll
      for (int m = 8; m; m >>= 1) rs += __shfl_xor(rs, m, 16);
      lrow[i] = lrow[i] * f + rs;
      mrow[i] = mnew;
      pr0[i] = p0; pr1[i] = p1; fsc[i] = f;
    }
#pragma unroll
    for (int j = 0; j < 8; ++j)
#pragma unroll
      for (int i = 0; i < 8; ++i) acc[j][i] *= fsc[i];

    // stage P (16x32 row-major bf16) in this wave's LDS slice. DS ops from one
    // wave execute in order (ISA 7.3), so the A-layout reload below observes
    // these stores; the compiler's own dscnt wait covers data availability.
    {
      int rb = (lane < 16) ? 0 : 8;
#pragma unroll
      for (int i = 0; i < 8; ++i) {
        smem.p[wave][(rb + i) * 32 + colK]      = f2bf(pr0[i]);
        smem.p[wave][(rb + i) * 32 + 16 + colK] = f2bf(pr1[i]);
      }
    }

    // reload P with the A-fragment interleave, batch the 8 V-fragments,
    // then burst the 8 PV WMMAs.
    v16bf pf;
    {
      const unsigned short* Pr =
          &smem.p[wave][(lane & 15) * 32 + ((lane < 16) ? 0 : 8)];
      pf = load_frag_pair(Pr, Pr + 16);
    }
    v16bf bv[8];
#pragma unroll
    for (int j = 0; j < 8; ++j) {
      const unsigned short* Bv = vb + (size_t)(j * 16 + colK) * 32 + bko;
      bv[j] = load_frag_pair(Bv, Bv + 16);
    }
    SCHED_FENCE();   // keep the V-fragment ds_loads ahead of the WMMA burst
#pragma unroll
    for (int j = 0; j < 8; ++j) acc[j] = wmma_bf16(pf, bv[j], acc[j]);

    __syncthreads();   // everyone done with buf before it is overwritten
  }

  // epilogue: divide by row sums, write bf16 [B, T, HIDDEN]
  const int b = bh / NHEADS, h = bh % NHEADS;
#pragma unroll
  for (int j = 0; j < 8; ++j) {
    int col = h * HDIM + j * 16 + (lane & 15);
#pragma unroll
    for (int i = 0; i < 8; ++i) {
      int row = q0 + i + ((lane < 16) ? 0 : 8);
      Out[((size_t)(b * SEQ) + row) * HIDDEN + col] = f2bf(acc[j][i] / lrow[i]);
    }
  }
}

// ---------------- launch ----------------
extern "C" void kernel_launch(void* const* d_in, const int* in_sizes, int n_in,
                              void* d_out, int out_size, void* d_ws,
                              size_t ws_size, hipStream_t stream) {
  (void)in_sizes; (void)n_in; (void)out_size; (void)ws_size;
  const float* x  = (const float*)d_in[0];
  const float* wq = (const float*)d_in[1];
  const float* wk = (const float*)d_in[2];
  const float* wv = (const float*)d_in[3];
  const float* wo = (const float*)d_in[4];

  char* ws = (char*)d_ws;
  constexpr size_t SZ_X = (size_t)NROWS * HIDDEN * 2;       // 20,971,520 B
  constexpr size_t SZ_W = (size_t)HIDDEN * HIDDEN * 2;      //  3,276,800 B
  unsigned short* Xb = (unsigned short*)(ws);
  unsigned short* Wq = (unsigned short*)(ws + SZ_X);
  unsigned short* Wk = (unsigned short*)(ws + SZ_X + 1 * SZ_W);
  unsigned short* Wv = (unsigned short*)(ws + SZ_X + 2 * SZ_W);
  unsigned short* Wo = (unsigned short*)(ws + SZ_X + 3 * SZ_W);
  size_t base = SZ_X + 4 * SZ_W;
  unsigned short* Qr = (unsigned short*)(ws + base + 0 * SZ_X);
  unsigned short* Kr = (unsigned short*)(ws + base + 1 * SZ_X);
  unsigned short* Vr = (unsigned short*)(ws + base + 2 * SZ_X);
  unsigned short* Qb = (unsigned short*)(ws + base + 3 * SZ_X);
  unsigned short* Kb = (unsigned short*)(ws + base + 4 * SZ_X);
  unsigned short* Vt = (unsigned short*)(ws + base + 5 * SZ_X);
  unsigned short* AttnOut = Xb;   // Xb dead after the QKV GEMMs -> reuse

  const int nx = NROWS * HIDDEN;
  const int nw = HIDDEN * HIDDEN;
  k_f32_to_bf16<<<(nx + 255) / 256, 256, 0, stream>>>(x, Xb, nx);
  k_f32_to_bf16<<<(nw + 255) / 256, 256, 0, stream>>>(wq, Wq, nw);
  k_f32_to_bf16<<<(nw + 255) / 256, 256, 0, stream>>>(wk, Wk, nw);
  k_f32_to_bf16<<<(nw + 255) / 256, 256, 0, stream>>>(wv, Wv, nw);
  k_f32_to_bf16<<<(nw + 255) / 256, 256, 0, stream>>>(wo, Wo, nw);

  dim3 gg(NROWS / 64, HIDDEN / 128);
  k_gemm_bf16<unsigned short><<<gg, 256, 0, stream>>>(Xb, Wq, Qr, NROWS, HIDDEN, HIDDEN);
  k_gemm_bf16<unsigned short><<<gg, 256, 0, stream>>>(Xb, Wk, Kr, NROWS, HIDDEN, HIDDEN);
  k_gemm_bf16<unsigned short><<<gg, 256, 0, stream>>>(Xb, Wv, Vr, NROWS, HIDDEN, HIDDEN);

  k_rope_rms<<<BATCH * SEQ * NHEADS, 128, 0, stream>>>(Qr, Kr, Vr, Qb, Kb, Vt);

  dim3 ga(SEQ / 128, BATCH * NHEADS);
  k_attn<<<ga, 256, 0, stream>>>(Qb, Kb, Vt, AttnOut);

  k_gemm_bf16<float><<<gg, 256, 0, stream>>>(AttnOut, Wo, (float*)d_out,
                                             NROWS, HIDDEN, HIDDEN);
}
